// LuongGCN_28441273434411
// MI455X (gfx1250) — matrix operand order repaired and checked by reading, not compiled, not run on hardware
//
#include <hip/hip_runtime.h>
#include <hip/hip_bf16.h>

// ---------------------------------------------------------------------------
// LuongGCN for MI455X (gfx1250): bf16 WMMA GEMMs + L2-resident edge scatter.
// Problem constants (from reference): N=50000, D=256, K=3, E=600000,
// H1=128, H2=64, 3 layers, eps=1e-5, leaky slope=0.01.
// ---------------------------------------------------------------------------

typedef __attribute__((ext_vector_type(16))) __bf16 v16bf;
typedef __attribute__((ext_vector_type(8)))  float  v8f;

union FragBF { unsigned u[8]; v16bf v; };
union FragF  { float    f[8]; v8f   v; };

__device__ __forceinline__ __bf16 f2bf(float f) {
    union { float f; unsigned u; } in; in.f = f;
    unsigned u = in.u;
    unsigned r = u + 0x7FFFu + ((u >> 16) & 1u);   // round-to-nearest-even
    unsigned short h = (unsigned short)(r >> 16);
    union { unsigned short s; __bf16 b; } out; out.s = h;
    return out.b;
}

// ------------------------- BatchNorm (training fwd) -------------------------

__global__ void bn_reduce_kernel(const float* __restrict__ x, float* __restrict__ sums,
                                 int N, int D) {
    int d = threadIdx.x;                       // D == blockDim.x == 256
    float s = 0.f, s2 = 0.f;
    for (int n = blockIdx.x; n < N; n += gridDim.x) {
        float v = x[(size_t)n * D + d];
        s += v; s2 += v * v;
    }
    atomicAdd(&sums[d], s);
    atomicAdd(&sums[D + d], s2);
}

__global__ void bn_finalize_kernel(const float* __restrict__ sums,
                                   const float* __restrict__ gamma,
                                   const float* __restrict__ beta,
                                   float* __restrict__ bnp, int N, int D) {
    int d = threadIdx.x;
    float mean = sums[d] / (float)N;
    float var  = sums[D + d] / (float)N - mean * mean;
    float s    = gamma[d] * rsqrtf(var + 1e-5f);
    bnp[d]     = s;
    bnp[D + d] = beta[d] - mean * s;           // y = x*s + shift
}

__global__ void bn_apply_kernel(const float* __restrict__ xin,
                                const float* __restrict__ bnp,
                                float* __restrict__ xout, __bf16* __restrict__ xb,
                                long total, int D) {
    long i = (long)blockIdx.x * 256 + threadIdx.x;
    if (i >= total) return;
    int d = (int)(i & (D - 1));
    float v = xin[i] * bnp[d] + bnp[D + d];
    xout[i] = v;
    xb[i]   = f2bf(v);
}

// ------------------- Edge norm precompute (layer-invariant) -----------------

__global__ void deg_kernel(const int* __restrict__ eidx, const float* __restrict__ ew,
                           float* __restrict__ deg, int E, int N, int K) {
    long tid = (long)blockIdx.x * 256 + threadIdx.x;
    if (tid >= (long)K * E) return;
    int k = (int)(tid / E), e = (int)(tid % E);
    int d = eidx[(size_t)k * 2 * E + E + e];   // dst
    atomicAdd(&deg[(size_t)k * N + d], ew[tid]);
}

__global__ void norm_kernel(const int* __restrict__ eidx, const float* __restrict__ ew,
                            const float* __restrict__ deg, float* __restrict__ nrm,
                            int E, int N, int K) {
    long tid = (long)blockIdx.x * 256 + threadIdx.x;
    if (tid >= (long)K * E) return;
    int k = (int)(tid / E), e = (int)(tid % E);
    const int* base = eidx + (size_t)k * 2 * E;
    int s = base[e], d = base[E + e];
    float ds = deg[(size_t)k * N + s], dd = deg[(size_t)k * N + d];
    float is = ds > 0.f ? rsqrtf(ds) : 0.f;
    float id = dd > 0.f ? rsqrtf(dd) : 0.f;
    nrm[tid] = ew[tid] * is * id;
}

// ------------------------------- Aggregation --------------------------------

__global__ void fill_bias_kernel(float* __restrict__ xs, const float* __restrict__ b,
                                 long total, int D) {
    long i = (long)blockIdx.x * 256 + threadIdx.x;
    if (i < total) xs[i] = b[(int)(i & (D - 1))];
}

// One block (D=256 threads) sweeps a chunk of edges; thread t owns feature t.
__global__ void aggregate_kernel(const float* __restrict__ h, const int* __restrict__ src,
                                 const int* __restrict__ dst, const float* __restrict__ nrm,
                                 float* __restrict__ xs, int E, int D, int epb) {
    int t  = threadIdx.x;
    int e0 = blockIdx.x * epb;
    int e1 = e0 + epb; if (e1 > E) e1 = E;
    for (int e = e0; e < e1; ++e) {
        float w = nrm[e];
        int   s = src[e];
        if (e + 1 < e1)                        // emits global_prefetch_b8
            __builtin_prefetch(&h[(size_t)src[e + 1] * D + t], 0, 0);
        if (w != 0.f) {
            float v = h[(size_t)s * D + t] * w;
            atomicAdd(&xs[(size_t)dst[e] * D + t], v);  // L2-resident scatter
        }
    }
}

// ------------------- Attention mix (K=3 Luong dot + softmax) ----------------

__global__ void attn_kernel(float* __restrict__ x, const float* __restrict__ xs,
                            __bf16* __restrict__ xb, int N, int D) {
    __shared__ float sred[256];
    __shared__ float sc[3];
    int n = blockIdx.x, t = threadIdx.x;
    size_t base = (size_t)n * D + t;
    size_t ND   = (size_t)N * D;
    float xv = x[base];
    float v0 = xs[base];
    float v1 = xs[ND + base];
    float v2 = xs[2 * ND + base];
    float pp[3] = { xv * v0, xv * v1, xv * v2 };
#pragma unroll
    for (int k = 0; k < 3; ++k) {
        sred[t] = pp[k];
        __syncthreads();
        for (int s = 128; s > 0; s >>= 1) {
            if (t < s) sred[t] += sred[t + s];
            __syncthreads();
        }
        if (t == 0) sc[k] = sred[0];
        __syncthreads();
    }
    float m  = fmaxf(sc[0], fmaxf(sc[1], sc[2]));
    float e0 = __expf(sc[0] - m), e1 = __expf(sc[1] - m), e2 = __expf(sc[2] - m);
    float inv = 1.f / (e0 + e1 + e2);
    float o = fmaxf(0.f, (e0 * v0 + e1 * v1 + e2 * v2) * inv);
    x[base]  = o;
    xb[base] = f2bf(o);
}

// ----------------------------- Weight transpose -----------------------------

// W is Kd x Nn row-major (f32); Wt is Nn x Kd row-major (bf16) = W^T.
__global__ void convT_kernel(const float* __restrict__ W, __bf16* __restrict__ Wt,
                             int Kd, int Nn) {
    int idx = blockIdx.x * 256 + threadIdx.x;
    if (idx >= Kd * Nn) return;
    int n = idx / Kd, kk = idx % Kd;
    Wt[idx] = f2bf(W[(size_t)kk * Nn + n]);
}

// ------------------------------ WMMA bf16 GEMM ------------------------------
// C[M,Nn] = A[M,Kd] (bf16, row-major) * Bt[Nn,Kd]^T (bf16) [+ bias] [leaky]
// One wave -> 16x64 output strip (4x v_wmma_f32_16x16x32_bf16 per K-step).
// act: 0 = none, 1 = leaky-relu(0.01). Cb: optional bf16 mirror of C.

__global__ void wmma_gemm_kernel(const __bf16* __restrict__ A, const __bf16* __restrict__ Bt,
                                 const float* __restrict__ bias, float* __restrict__ C,
                                 __bf16* __restrict__ Cb,
                                 int M, int Kd, int Nn, int act) {
    int lane = threadIdx.x & 31;
    int wid  = (int)((blockIdx.x * (long)blockDim.x + threadIdx.x) >> 5);
    int nStrips = Nn >> 6;                 // strips of 64 columns
    int totalW  = (M >> 4) * nStrips;
    if (wid >= totalW) return;             // wave-uniform: EXEC stays all-ones

    int mTile = wid / nStrips;
    int strip = wid - mTile * nStrips;
    int half  = lane >> 4;                 // lane half selects K sub-group
    int l16   = lane & 15;

    FragF acc[4];
#pragma unroll
    for (int j = 0; j < 4; ++j)
#pragma unroll
        for (int r = 0; r < 8; ++r) acc[j].f[r] = 0.f;

    const __bf16* Arow = A + (size_t)(mTile * 16 + l16) * Kd;

    for (int kb = 0; kb < Kd; kb += 32) {
        // A fragment: 16x32 bf16, VGPR v holds K = half*16 + 2v(+1)
        FragBF a;
#pragma unroll
        for (int v = 0; v < 8; ++v) {
            int ko = half * 16 + 2 * v;
            a.u[v] = *reinterpret_cast<const unsigned*>(Arow + kb + ko);
        }
#pragma unroll
        for (int j = 0; j < 4; ++j) {
            // B fragment: 32x16 bf16, lane = N, VGPR v holds K = half*16 + 2v(+1)
            const __bf16* Brow = Bt + (size_t)(strip * 64 + j * 16 + l16) * Kd
                                    + kb + half * 16;
            FragBF b;
#pragma unroll
            for (int v = 0; v < 8; ++v)
                b.u[v] = *reinterpret_cast<const unsigned*>(Brow + 2 * v);
            acc[j].v = __builtin_amdgcn_wmma_f32_16x16x32_bf16(
                false, a.v, false, b.v, (short)0, acc[j].v, false, false);
        }
    }

    // Epilogue: C layout — lanes 0-15: M=r, lanes 16-31: M=8+r; N = lane&15.
#pragma unroll
    for (int j = 0; j < 4; ++j) {
        int col = strip * 64 + j * 16 + l16;
        float bv = bias ? bias[col] : 0.f;
#pragma unroll
        for (int r = 0; r < 8; ++r) {
            int row = mTile * 16 + half * 8 + r;
            float v = acc[j].f[r] + bv;
            if (act == 1) v = (v > 0.f) ? v : 0.01f * v;
            C[(size_t)row * Nn + col] = v;
            if (Cb) Cb[(size_t)row * Nn + col] = f2bf(v);
        }
    }
}

// --------------------------------- Launch -----------------------------------

extern "C" void kernel_launch(void* const* d_in, const int* in_sizes, int n_in,
                              void* d_out, int out_size, void* d_ws, size_t ws_size,
                              hipStream_t stream) {
    (void)in_sizes; (void)n_in; (void)out_size; (void)ws_size;

    const float* x_in  = (const float*)d_in[0];
    const int*   eidx  = (const int*)  d_in[1];   // [K,2,E]
    const float* ew    = (const float*)d_in[2];   // [K,E]
    const float* gamma = (const float*)d_in[3];
    const float* beta  = (const float*)d_in[4];
    const float* gcnw  = (const float*)d_in[5];   // [3,D,D]
    const float* gcnb  = (const float*)d_in[6];   // [3,D]
    const float* w1    = (const float*)d_in[7];   // [D,H1]
    const float* b1    = (const float*)d_in[8];
    const float* w2    = (const float*)d_in[9];   // [H1,H2]
    const float* b2    = (const float*)d_in[10];
    float* out = (float*)d_out;                   // [N,H2] f32

    const int N = 50000, D = 256, K = 3, E = 600000, H1 = 128, H2 = 64;
    const long ND = (long)N * D;

    // Workspace carve-out (~328 MB total)
    char* ws = (char*)d_ws;
    size_t off = 0;
    auto carve = [&](size_t bytes) -> char* {
        char* p = ws + off;
        off = (off + bytes + 255) & ~(size_t)255;
        return p;
    };
    float*  xcur = (float*) carve((size_t)ND * 4);
    __bf16* xb   = (__bf16*)carve((size_t)ND * 2);
    float*  h    = (float*) carve((size_t)ND * 4);
    float*  xs   = (float*) carve((size_t)K * ND * 4);
    float*  deg  = (float*) carve((size_t)K * N * 4);
    float*  nrm  = (float*) carve((size_t)K * E * 4);
    __bf16* Wt   = (__bf16*)carve((size_t)D * D * 2);
    __bf16* Wt1  = (__bf16*)carve((size_t)D * H1 * 2);
    __bf16* Wt2  = (__bf16*)carve((size_t)H1 * H2 * 2);
    float*  h1   = (float*) carve((size_t)N * H1 * 4);
    __bf16* h1b  = (__bf16*)carve((size_t)N * H1 * 2);
    float*  sums = (float*) carve(2 * D * 4);
    float*  bnp  = (float*) carve(2 * D * 4);

    auto cdiv = [](long a, long b) { return (int)((a + b - 1) / b); };

    // --- BatchNorm (batch stats), fused bf16 conversion ---
    hipMemsetAsync(sums, 0, 2 * D * sizeof(float), stream);
    bn_reduce_kernel<<<512, D, 0, stream>>>(x_in, sums, N, D);
    bn_finalize_kernel<<<1, D, 0, stream>>>(sums, gamma, beta, bnp, N, D);
    bn_apply_kernel<<<cdiv(ND, 256), 256, 0, stream>>>(x_in, bnp, xcur, xb, ND, D);

    // --- Layer-invariant edge normalization ---
    hipMemsetAsync(deg, 0, (size_t)K * N * sizeof(float), stream);
    deg_kernel<<<cdiv((long)K * E, 256), 256, 0, stream>>>(eidx, ew, deg, E, N, K);
    norm_kernel<<<cdiv((long)K * E, 256), 256, 0, stream>>>(eidx, ew, deg, nrm, E, N, K);

    // --- GCN layers ---
    for (int l = 0; l < 3; ++l) {
        convT_kernel<<<cdiv((long)D * D, 256), 256, 0, stream>>>(
            gcnw + (size_t)l * D * D, Wt, D, D);
        {
            long waves = (long)(N / 16) * (D / 64);
            wmma_gemm_kernel<<<cdiv(waves * 32, 256), 256, 0, stream>>>(
                xb, Wt, nullptr, h, nullptr, N, D, D, 0);
        }
        for (int k = 0; k < K; ++k) {
            fill_bias_kernel<<<cdiv(ND, 256), 256, 0, stream>>>(
                xs + (size_t)k * ND, gcnb + (size_t)l * D, ND, D);
            const int* src = eidx + (size_t)k * 2 * E;
            const int* dst = src + E;
            aggregate_kernel<<<cdiv(E, 32), D, 0, stream>>>(
                h, src, dst, nrm + (size_t)k * E, xs + (size_t)k * ND, E, D, 32);
        }
        attn_kernel<<<N, D, 0, stream>>>(xcur, xs, xb, N, D);
    }

    // --- Projection head ---
    convT_kernel<<<cdiv((long)D * H1, 256), 256, 0, stream>>>(w1, Wt1, D, H1);
    {
        long waves = (long)(N / 16) * (H1 / 64);
        wmma_gemm_kernel<<<cdiv(waves * 32, 256), 256, 0, stream>>>(
            xb, Wt1, b1, h1, h1b, N, D, H1, 1);
    }
    convT_kernel<<<cdiv((long)H1 * H2, 256), 256, 0, stream>>>(w2, Wt2, H1, H2);
    {
        long waves = (long)(N / 16) * (H2 / 64);
        wmma_gemm_kernel<<<cdiv(waves * 32, 256), 256, 0, stream>>>(
            h1b, Wt2, b2, out, nullptr, N, H1, H2, 1);
    }
}
